// MainAssistAttention_88553635709444
// MI455X (gfx1250) — compile-verified, matrix operand order checked by reading, not run on hardware
//
#include <hip/hip_runtime.h>
#include <hip/hip_bf16.h>
#include <math.h>

typedef __bf16 bf16_t;
typedef __attribute__((ext_vector_type(16))) __bf16 v16bf;
typedef __attribute__((ext_vector_type(8)))  float  v8f;

#define NDIM 4
#define BDIM 4
#define CDIM 64
#define HDIM 256
#define WDIM 256
#define HW   65536

// ---------------------------------------------------------------------------
// CDNA5 async global->LDS path (ASYNCcnt), with sync fallback
// ---------------------------------------------------------------------------
#if defined(__has_builtin)
#  if __has_builtin(__builtin_amdgcn_global_load_async_to_lds_b128)
#    define HAVE_ASYNC_LDS 1
#  endif
#  if __has_builtin(__builtin_amdgcn_s_wait_asynccnt)
#    define HAVE_WAIT_ASYNC_BUILTIN 1
#  endif
#endif

typedef int v4i32 __attribute__((vector_size(16)));

template <int N>
__device__ inline void wait_async() {
#if defined(HAVE_ASYNC_LDS)
#  if defined(HAVE_WAIT_ASYNC_BUILTIN)
  __builtin_amdgcn_s_wait_asynccnt(N);
#  else
  asm volatile("s_wait_asynccnt %0" ::"n"(N) : "memory");
#  endif
#endif
}

__device__ inline void async_b128(bf16_t* lds_dst, const bf16_t* gsrc) {
#if defined(HAVE_ASYNC_LDS)
  __builtin_amdgcn_global_load_async_to_lds_b128(
      (__attribute__((address_space(1))) v4i32*)gsrc,
      (__attribute__((address_space(3))) v4i32*)lds_dst, 0, 0);
#else
  *(uint4*)lds_dst = *(const uint4*)gsrc;
#endif
}

// Contiguous tile copy, BYTES must be a multiple of 4096 (256 thr * 16B).
// Issues BYTES/4096 async b128 ops per thread.
template <int BYTES>
__device__ inline void tile_load(bf16_t* dst, const bf16_t* src) {
  int tid = threadIdx.x;
#pragma unroll
  for (int i = 0; i < BYTES / 4096; ++i) {
    int idx = i * 256 + tid;
    async_b128(dst + idx * 8, src + idx * 8);
  }
}

// K w-column tile: Ktile[g][0..31] = K[g][wofs + 0..31]; one 64B row per
// thread (4 async b128 per thread).
__device__ inline void ktile_load(bf16_t* dst, const bf16_t* Kg, int wofs) {
  int tid = threadIdx.x;
  const bf16_t* s = Kg + (size_t)tid * 256 + wofs;
  bf16_t* d = dst + tid * 32;
#pragma unroll
  for (int j = 0; j < 4; ++j) async_b128(d + j * 8, s + j * 8);
}

// ---------------------------------------------------------------------------
// WMMA fragment helpers (CDNA5 wave32 layouts, 16x16x32 bf16)
// ---------------------------------------------------------------------------

// Element (r, k) from a row-major [r][k] tile (r = lane&15).
// A-matrix from MxK storage, or B-matrix from NxK storage (B transposed).
__device__ inline v16bf frag_rowk(const bf16_t* p, int ld) {
  int lane = threadIdx.x & 31;
  int r  = lane & 15;
  int kh = (lane >> 4) << 3;
  v16bf f;
#pragma unroll
  for (int e = 0; e < 16; ++e) {
    int j = e >> 1;
    int k = ((j < 4) ? (j << 1) : ((j << 1) + 8)) + (e & 1) + kh;
    f[e] = p[r * ld + k];
  }
  return f;
}

// Element (k, r) from a row-major [k][r] tile (r = lane&15).
// B-matrix from KxN storage, or A-matrix accessed transposed.
__device__ inline v16bf frag_krow(const bf16_t* p, int ld) {
  int lane = threadIdx.x & 31;
  int r  = lane & 15;
  int kh = (lane >> 4) << 3;
  v16bf f;
#pragma unroll
  for (int e = 0; e < 16; ++e) {
    int j = e >> 1;
    int k = ((j < 4) ? (j << 1) : ((j << 1) + 8)) + (e & 1) + kh;
    f[e] = p[k * ld + r];
  }
  return f;
}

__device__ inline v8f wmma_bf16(v16bf a, v16bf b, v8f c) {
  return __builtin_amdgcn_wmma_f32_16x16x32_bf16(
      false, a, false, b, (short)0, c, false, false);
}

// C/D layout: VGPR r holds M = r + 8*(lane>>4); N = lane&15.
__device__ inline void store_tile_f32(float* p, int ld, v8f acc) {
  int lane = threadIdx.x & 31;
  int n  = lane & 15;
  int mh = (lane >> 4) << 3;
#pragma unroll
  for (int r = 0; r < 8; ++r) p[(r + mh) * ld + n] = acc[r];
}

__device__ inline void atomic_add_tile(float* p, int ld, v8f acc) {
  int lane = threadIdx.x & 31;
  int n  = lane & 15;
  int mh = (lane >> 4) << 3;
#pragma unroll
  for (int r = 0; r < 8; ++r) unsafeAtomicAdd(&p[(r + mh) * ld + n], acc[r]);
}

// Row softmax over a 32x256 fp32 block (8 threads/row), bf16 probs -> P.
__device__ inline void softmax_rows32(float* Ss, bf16_t* P, float* red, float scale) {
  int tid = threadIdx.x;
  int row = tid >> 3, sub = tid & 7;
  float* srow = Ss + row * 256 + sub * 32;
  float lmax = -1e30f;
#pragma unroll
  for (int j = 0; j < 32; ++j) lmax = fmaxf(lmax, srow[j]);
  red[tid] = lmax;
  __syncthreads();
  float rmax = -1e30f;
#pragma unroll
  for (int s = 0; s < 8; ++s) rmax = fmaxf(rmax, red[(row << 3) + s]);
  __syncthreads();
  float lsum = 0.f;
#pragma unroll
  for (int j = 0; j < 32; ++j) {
    float e = __expf((srow[j] - rmax) * scale);
    srow[j] = e;
    lsum += e;
  }
  red[tid] = lsum;
  __syncthreads();
  float rsum = 0.f;
#pragma unroll
  for (int s = 0; s < 8; ++s) rsum += red[(row << 3) + s];
  float inv = 1.f / rsum;
  bf16_t* prow = P + row * 256 + sub * 32;
#pragma unroll
  for (int j = 0; j < 32; ++j) prow[j] = (bf16_t)(srow[j] * inv);
}

// ---------------------------------------------------------------------------
// out = main_input
// ---------------------------------------------------------------------------
__global__ void copy_kernel(const float4* __restrict__ src, float4* __restrict__ dst, int n4) {
  int i = blockIdx.x * blockDim.x + threadIdx.x;
  int stride = gridDim.x * blockDim.x;
  for (; i < n4; i += stride) dst[i] = src[i];
}

// ---------------------------------------------------------------------------
// 1x1 conv projection: out[batch, co, hw] = sum_ci w[co,ci]*x[batch,ci,hw]+b[co]
// ---------------------------------------------------------------------------
__global__ __launch_bounds__(256) void proj_kernel(
    const float* __restrict__ x, const float* __restrict__ w,
    const float* __restrict__ bias, bf16_t* __restrict__ out, int npix) {
  __shared__ float wsh[CDIM * CDIM];
  __shared__ float bsh[CDIM];
  int tid = threadIdx.x;
  for (int i = tid; i < CDIM * CDIM; i += 256) wsh[i] = w[i];
  if (tid < CDIM) bsh[tid] = bias[tid];
  __syncthreads();

  int p = blockIdx.x * 256 + tid;
  if (p >= npix) return;
  int batch = p >> 16;  // HW = 65536
  int hw    = p & 65535;

  const float* xp = x + (size_t)batch * CDIM * HW + hw;
  float xr[CDIM];
#pragma unroll
  for (int ci = 0; ci < CDIM; ++ci) xr[ci] = xp[(size_t)ci * HW];

  bf16_t* op = out + (size_t)batch * CDIM * HW + hw;
#pragma unroll 4
  for (int co = 0; co < CDIM; ++co) {
    float acc = bsh[co];
    const float* wr = &wsh[co * CDIM];
#pragma unroll
    for (int ci = 0; ci < CDIM; ++ci) acc = fmaf(wr[ci], xr[ci], acc);
    op[(size_t)co * HW] = (bf16_t)acc;
  }
}

// ---------------------------------------------------------------------------
// Time attention: per (n,b,c): S = Q K^T / 16 ; P = softmax_rows(S);
// out += P V.  Async double-buffered K/V tile streaming.
// ---------------------------------------------------------------------------
__global__ __launch_bounds__(256) void time_attn_kernel(
    const bf16_t* __restrict__ Qb, const bf16_t* __restrict__ Kb,
    const bf16_t* __restrict__ Vb, float* __restrict__ out) {
  __shared__ __align__(16) bf16_t Qs[32 * 256];       // Q rows, later P
  __shared__ __align__(16) bf16_t Ks[2][256 * 32];    // K w-column chunk [g][wl]
  __shared__ __align__(16) bf16_t Vs[2][32 * 256];    // V g-chunk rows
  __shared__ __align__(16) float  Ss[32 * 256];
  __shared__ float red[256];

  int idx = blockIdx.x;
  int BC = BDIM * CDIM;
  int n  = idx / BC;
  int bc = idx - n * BC;
  const bf16_t* Qg = Qb + (size_t)bc * HW;
  const bf16_t* Vg = Vb + (size_t)bc * HW;
  const bf16_t* Kg = Kb + ((size_t)n * BC + bc) * HW;
  float* outg = out + (size_t)bc * HW;

  int tid  = threadIdx.x;
  int wave = tid >> 5;

  for (int hb = 0; hb < 8; ++hb) {
    __syncthreads();
    // prefetch Q block (4/thr) then first K tile (4/thr)
    tile_load<16384>(Qs, Qg + hb * 32 * 256);
    ktile_load(Ks[0], Kg, 0);
    wait_async<4>();          // Q complete (in-order); Ks[0] may be in flight
    __syncthreads();

    // ---- S = Q K^T (k-dim = w streamed in 32-col K tiles) ----
    v8f acc[4] = {};
    for (int kk = 0; kk < 8; ++kk) {
      __syncthreads();        // readers of victim buffer are done
      if (kk < 7) {
        ktile_load(Ks[(kk + 1) & 1], Kg, (kk + 1) * 32);
        wait_async<4>();      // tile kk complete
      } else {
        wait_async<0>();
      }
      __syncthreads();        // tile kk visible to all waves
      const bf16_t* Kcur = Ks[kk & 1];
#pragma unroll
      for (int t = 0; t < 4; ++t) {
        int tile = wave * 4 + t;  // 2 row-tiles x 16 col-tiles
        int rt = tile >> 4, ct = tile & 15;
        v16bf a = frag_rowk(Qs + rt * 16 * 256 + kk * 32, 256);  // A[m=h,k=w]
        v16bf b = frag_rowk(Kcur + ct * 16 * 32, 32);            // B[k=w,n=g]
        acc[t] = wmma_bf16(a, b, acc[t]);
      }
    }
    __syncthreads();
#pragma unroll
    for (int t = 0; t < 4; ++t) {
      int tile = wave * 4 + t;
      int rt = tile >> 4, ct = tile & 15;
      store_tile_f32(Ss + rt * 16 * 256 + ct * 16, 256, acc[t]);
    }
    __syncthreads();

    softmax_rows32(Ss, Qs, red, 0.0625f);   // 1/sqrt(256); P -> Qs
    __syncthreads();

    // ---- O = P V (k-dim = g, V streamed in 32-row chunks) ----
    tile_load<16384>(Vs[0], Vg);
    v8f oacc[4] = {};
    for (int kk = 0; kk < 8; ++kk) {
      __syncthreads();
      if (kk < 7) {
        tile_load<16384>(Vs[(kk + 1) & 1], Vg + (kk + 1) * 32 * 256);
        wait_async<4>();
      } else {
        wait_async<0>();
      }
      __syncthreads();
      const bf16_t* Vcur = Vs[kk & 1];
#pragma unroll
      for (int t = 0; t < 4; ++t) {
        int tile = wave * 4 + t;
        int rt = tile >> 4, ct = tile & 15;
        v16bf a = frag_rowk(Qs + rt * 16 * 256 + kk * 32, 256);  // A[m=h,k=g]=P
        v16bf b = frag_krow(Vcur + ct * 16, 256);                // B[k=g,n=w]=V
        oacc[t] = wmma_bf16(a, b, oacc[t]);
      }
    }
#pragma unroll
    for (int t = 0; t < 4; ++t) {
      int tile = wave * 4 + t;
      int rt = tile >> 4, ct = tile & 15;
      atomic_add_tile(outg + (hb * 32 + rt * 16) * 256 + ct * 16, 256, oacc[t]);
    }
  }
}

// ---------------------------------------------------------------------------
// Space attention: per (n,b,c): S = Q^T K / 16 ; P = softmax_rows(S);
// out += V[:, w-block] P_block  (k=32 per w-block, atomics accumulate)
// ---------------------------------------------------------------------------
__global__ __launch_bounds__(256) void space_attn_kernel(
    const bf16_t* __restrict__ Qb, const bf16_t* __restrict__ Kb,
    const bf16_t* __restrict__ Vb, float* __restrict__ out) {
  __shared__ __align__(16) bf16_t Qt[2][32 * 256];   // Q h-chunk rows
  __shared__ __align__(16) bf16_t Kt[2][32 * 256];   // K h-chunk rows
  __shared__ __align__(16) bf16_t Pt[32 * 256];      // P block (bf16)
  __shared__ __align__(16) bf16_t Vt[2][32 * 256];   // V h-chunk rows
  __shared__ __align__(16) float  Ss[32 * 256];
  __shared__ float red[256];

  int idx = blockIdx.x;
  int BC = BDIM * CDIM;
  int n  = idx / BC;
  int bc = idx - n * BC;
  const bf16_t* Qg = Qb + (size_t)bc * HW;
  const bf16_t* Vg = Vb + (size_t)bc * HW;
  const bf16_t* Kg = Kb + ((size_t)n * BC + bc) * HW;
  float* outg = out + (size_t)bc * HW;

  int tid  = threadIdx.x;
  int wave = tid >> 5;

  for (int wb = 0; wb < 8; ++wb) {
    __syncthreads();
    tile_load<16384>(Qt[0], Qg);
    tile_load<16384>(Kt[0], Kg);

    // ---- S = Q^T K for w-rows [wb*32,+32), k-dim = h streamed ----
    v8f acc[4] = {};
    for (int kk = 0; kk < 8; ++kk) {
      __syncthreads();
      if (kk < 7) {
        tile_load<16384>(Qt[(kk + 1) & 1], Qg + (kk + 1) * 32 * 256);
        tile_load<16384>(Kt[(kk + 1) & 1], Kg + (kk + 1) * 32 * 256);
        wait_async<8>();     // chunk kk (8/thr) complete
      } else {
        wait_async<0>();
      }
      __syncthreads();
      const bf16_t* Qcur = Qt[kk & 1];
      const bf16_t* Kcur = Kt[kk & 1];
#pragma unroll
      for (int t = 0; t < 4; ++t) {
        int tile = wave * 4 + t;
        int rt = tile >> 4, ct = tile & 15;
        v16bf a = frag_krow(Qcur + wb * 32 + rt * 16, 256);  // A[m=w,k=h]=Q^T
        v16bf b = frag_krow(Kcur + ct * 16, 256);            // B[k=h,n=v]=K
        acc[t] = wmma_bf16(a, b, acc[t]);
      }
    }
    __syncthreads();
#pragma unroll
    for (int t = 0; t < 4; ++t) {
      int tile = wave * 4 + t;
      int rt = tile >> 4, ct = tile & 15;
      store_tile_f32(Ss + rt * 16 * 256 + ct * 16, 256, acc[t]);
    }
    __syncthreads();

    softmax_rows32(Ss, Pt, red, 0.0625f);   // 1/sqrt(256)
    __syncthreads();

    // ---- out[h, v] += sum_{w in block} V[h][w] * P[w][v]  (k = 32) ----
    tile_load<16384>(Vt[0], Vg);
    for (int hbk = 0; hbk < 8; ++hbk) {
      __syncthreads();
      if (hbk < 7) {
        tile_load<16384>(Vt[(hbk + 1) & 1], Vg + (hbk + 1) * 32 * 256);
        wait_async<4>();
      } else {
        wait_async<0>();
      }
      __syncthreads();
      const bf16_t* Vcur = Vt[hbk & 1];
#pragma unroll
      for (int t = 0; t < 4; ++t) {
        int tile = wave * 4 + t;
        int rt = tile >> 4, ct = tile & 15;
        v16bf a = frag_rowk(Vcur + rt * 16 * 256 + wb * 32, 256); // A[m=h,k=wl]
        v16bf b = frag_krow(Pt + ct * 16, 256);                   // B[k=wl,n=v]
        v8f o = {};
        o = wmma_bf16(a, b, o);
        atomic_add_tile(outg + (hbk * 32 + rt * 16) * 256 + ct * 16, 256, o);
      }
    }
  }
}

// ---------------------------------------------------------------------------
extern "C" void kernel_launch(void* const* d_in, const int* in_sizes, int n_in,
                              void* d_out, int out_size, void* d_ws, size_t ws_size,
                              hipStream_t stream) {
  const float* main_in = (const float*)d_in[0];
  const float* assist  = (const float*)d_in[1];
  const float* wq = (const float*)d_in[2];
  const float* bq = (const float*)d_in[3];
  const float* wk = (const float*)d_in[4];
  const float* bk = (const float*)d_in[5];
  const float* wv = (const float*)d_in[6];
  const float* bv = (const float*)d_in[7];
  float* out = (float*)d_out;

  const size_t mainElems = (size_t)BDIM * CDIM * HW;  // 16.78M
  bf16_t* Qbuf = (bf16_t*)d_ws;
  bf16_t* Vbuf = Qbuf + mainElems;
  bf16_t* Kbuf = Vbuf + mainElems;                    // N x mainElems

  copy_kernel<<<4096, 256, 0, stream>>>((const float4*)main_in, (float4*)out,
                                        (int)(mainElems / 4));
  proj_kernel<<<(BDIM * HW) / 256, 256, 0, stream>>>(main_in, wq, bq, Qbuf, BDIM * HW);
  proj_kernel<<<(BDIM * HW) / 256, 256, 0, stream>>>(main_in, wv, bv, Vbuf, BDIM * HW);
  proj_kernel<<<(NDIM * BDIM * HW) / 256, 256, 0, stream>>>(assist, wk, bk, Kbuf,
                                                            NDIM * BDIM * HW);
  time_attn_kernel<<<NDIM * BDIM * CDIM, 256, 0, stream>>>(Qbuf, Kbuf, Vbuf, out);
  space_attn_kernel<<<NDIM * BDIM * CDIM, 256, 0, stream>>>(Qbuf, Kbuf, Vbuf, out);
}
